// GroupAttention_49976239456532
// MI455X (gfx1250) — matrix operand order repaired and checked by reading, not compiled, not run on hardware
//
#include <hip/hip_runtime.h>
#include <hip/hip_bf16.h>

#define DIM   256
#define DIM3  768
#define HD    128
#define WAVES 4
#define RPW   16                 // rows per wave tile
#define RPB   (WAVES * RPW)      // rows per block

typedef __attribute__((ext_vector_type(16))) __bf16 v16bf;
typedef __attribute__((ext_vector_type(8)))  __bf16 v8bf;
typedef __attribute__((ext_vector_type(8)))  float  v8f;

// ---------------- weight fp32 -> bf16 conversion ----------------
__global__ void GA_wcvt_kernel(const float* __restrict__ qkv_w,
                               const float* __restrict__ out_w,
                               __bf16* __restrict__ wb) {
  int i = blockIdx.x * blockDim.x + threadIdx.x;
  const int NQ = DIM3 * DIM;
  const int NO = DIM * DIM;
  if (i < NQ)            wb[i] = (__bf16)qkv_w[i];
  else if (i < NQ + NO)  wb[i] = (__bf16)out_w[i - NQ];
}

// ---------------- fused LN + QKV + attention + out-proj ----------------
__global__ __launch_bounds__(WAVES * 32)
void GA_attn_kernel(const float* __restrict__ x,
                    const float* __restrict__ qkv_b,
                    const float* __restrict__ out_b,
                    const float* __restrict__ ln_g,
                    const float* __restrict__ ln_b,
                    const __bf16* __restrict__ qkv_wb,   // [768][256] bf16
                    const __bf16* __restrict__ out_wb,   // [256][256] bf16
                    float* __restrict__ out) {
  __shared__ __bf16 s_xn [WAVES][RPW * DIM];    // x-hat, later o           (8KB/wave)
  __shared__ __bf16 s_qkv[WAVES][RPW * DIM3];   // q|k|v, later f32 result  (24KB/wave)

  const int tid  = threadIdx.x;
  const int wave = tid >> 5;
  const int lane = tid & 31;
  const int r0   = blockIdx.x * RPB + wave * RPW;

  const int m     = lane & 15;   // row within tile (A-frag / LN row)
  const int khalf = lane >> 4;   // K-half selector (also column half for LN)
  const int ncol  = lane & 15;   // N column within a 16-wide tile (B/D frag)

  // ---------- Phase 1: LayerNorm (2 lanes per row, 128 cols each) ----------
  {
    const float* xr = x + (size_t)(r0 + m) * DIM + khalf * 128;
    float sum = 0.f, ss = 0.f;
    for (int j = 0; j < 128; j += 4) {
      float4 v = *(const float4*)(xr + j);
      sum += v.x + v.y + v.z + v.w;
      ss  += v.x * v.x + v.y * v.y + v.z * v.z + v.w * v.w;
    }
    sum += __shfl_xor(sum, 16, 32);
    ss  += __shfl_xor(ss,  16, 32);
    const float mu   = sum * (1.f / DIM);
    const float var  = ss * (1.f / DIM) - mu * mu;
    const float rstd = rsqrtf(var + 1e-5f);

    __bf16* xn = &s_xn[wave][m * DIM + khalf * 128];
    const float* gg = ln_g + khalf * 128;
    const float* bb = ln_b + khalf * 128;
    for (int j = 0; j < 128; j += 8) {
      float4 va = *(const float4*)(xr + j);
      float4 vb = *(const float4*)(xr + j + 4);
      float4 ga = *(const float4*)(gg + j);
      float4 gb = *(const float4*)(gg + j + 4);
      float4 ba = *(const float4*)(bb + j);
      float4 b2 = *(const float4*)(bb + j + 4);
      union { v8bf v; __bf16 e[8]; } pk;
      pk.e[0] = (__bf16)((va.x - mu) * rstd * ga.x + ba.x);
      pk.e[1] = (__bf16)((va.y - mu) * rstd * ga.y + ba.y);
      pk.e[2] = (__bf16)((va.z - mu) * rstd * ga.z + ba.z);
      pk.e[3] = (__bf16)((va.w - mu) * rstd * ga.w + ba.w);
      pk.e[4] = (__bf16)((vb.x - mu) * rstd * gb.x + b2.x);
      pk.e[5] = (__bf16)((vb.y - mu) * rstd * gb.y + b2.y);
      pk.e[6] = (__bf16)((vb.z - mu) * rstd * gb.z + b2.z);
      pk.e[7] = (__bf16)((vb.w - mu) * rstd * gb.w + b2.w);
      *(v8bf*)(xn + j) = pk.v;      // ds_store_b128
    }
  }
  __syncthreads();

  // ---------- Phase 2: QKV GEMM  (16x768 = 48 n-tiles x 8 k-steps) ----------
  for (int nt = 0; nt < 48; ++nt) {
    const int n = nt * 16 + ncol;
    const float bias = qkv_b[n];
    v8f acc;
#pragma unroll
    for (int r = 0; r < 8; ++r) acc[r] = bias;

    const __bf16* wrow = qkv_wb + (size_t)n * DIM;
    if (nt < 47) __builtin_prefetch(qkv_wb + (size_t)(n + 16) * DIM, 0, 1);

#pragma unroll
    for (int ks = 0; ks < 8; ++ks) {
      const int k0 = ks * 32;
      union { v16bf v; v8bf h[2]; } A, B;
      const __bf16* ap = &s_xn[wave][m * DIM + k0 + khalf * 8];
      A.h[0] = *(const v8bf*)ap;          // K = k0+khalf*8 .. +7
      A.h[1] = *(const v8bf*)(ap + 16);   // K = k0+16+khalf*8 .. +7
      const __bf16* bp = wrow + k0 + khalf * 8;
      B.h[0] = *(const v8bf*)bp;          // K = k0+khalf*8 .. +7
      B.h[1] = *(const v8bf*)(bp + 16);   // K = k0+16+khalf*8 .. +7
      acc = __builtin_amdgcn_wmma_f32_16x16x32_bf16(
          false, A.v, false, B.v, (short)0, acc, false, false);
    }
    // D layout: VGPR r, lanes<16 -> M=r, lanes>=16 -> M=8+r ; N = lane&15
    __bf16* qd = &s_qkv[wave][0];
#pragma unroll
    for (int r = 0; r < 8; ++r) {
      const int mr = (lane >> 4) * 8 + r;
      qd[mr * DIM3 + nt * 16 + ncol] = (__bf16)acc[r];
    }
  }
  __syncthreads();

  // ---------- Phase 3: per-group attention (vectorized LDS, in-lane softmax) ----------
  {
    const int gl = lane >> 3;         // local group 0..3
    const int h  = (lane >> 2) & 1;   // head
    const int ti = lane & 3;          // query token
    const int qrow = gl * 4 + ti;
    const __bf16* qbase = &s_qkv[wave][qrow * DIM3 + h * HD];
    const __bf16* kbase = &s_qkv[wave][(gl * 4) * DIM3 + DIM + h * HD];

    float s0 = 0.f, s1 = 0.f, s2 = 0.f, s3 = 0.f;
    for (int c = 0; c < HD; c += 8) {
      v8bf qv = *(const v8bf*)(qbase + c);
      v8bf k0 = *(const v8bf*)(kbase + c);
      v8bf k1 = *(const v8bf*)(kbase + DIM3 + c);
      v8bf k2 = *(const v8bf*)(kbase + 2 * DIM3 + c);
      v8bf k3 = *(const v8bf*)(kbase + 3 * DIM3 + c);
#pragma unroll
      for (int e = 0; e < 8; ++e) {
        const float q = (float)qv[e];
        s0 += q * (float)k0[e];
        s1 += q * (float)k1[e];
        s2 += q * (float)k2[e];
        s3 += q * (float)k3[e];
      }
    }
    const float sc = 0.08838834764831845f;  // 128^-0.5
    s0 *= sc; s1 *= sc; s2 *= sc; s3 *= sc;
    const float mx = fmaxf(fmaxf(s0, s1), fmaxf(s2, s3));
    float p0 = __expf(s0 - mx), p1 = __expf(s1 - mx);
    float p2 = __expf(s2 - mx), p3 = __expf(s3 - mx);
    const float inv = 1.f / (p0 + p1 + p2 + p3);
    p0 *= inv; p1 *= inv; p2 *= inv; p3 *= inv;

    const __bf16* vbase = &s_qkv[wave][(gl * 4) * DIM3 + 2 * DIM + h * HD];
    __bf16* op = &s_xn[wave][qrow * DIM + h * HD];    // overwrite x-hat with o
    for (int c = 0; c < HD; c += 8) {
      v8bf v0 = *(const v8bf*)(vbase + c);
      v8bf v1 = *(const v8bf*)(vbase + DIM3 + c);
      v8bf v2 = *(const v8bf*)(vbase + 2 * DIM3 + c);
      v8bf v3 = *(const v8bf*)(vbase + 3 * DIM3 + c);
      union { v8bf v; __bf16 e[8]; } o;
#pragma unroll
      for (int e = 0; e < 8; ++e)
        o.e[e] = (__bf16)(p0 * (float)v0[e] + p1 * (float)v1[e] +
                          p2 * (float)v2[e] + p3 * (float)v3[e]);
      *(v8bf*)(op + c) = o.v;       // ds_store_b128
    }
  }
  __syncthreads();

  // ---------- Phase 4: out-proj GEMM (f32 result staged in LDS) ----------
  float* obuf = (float*)&s_qkv[wave][0];      // 16 x 256 f32 (16KB, q/k/v dead)
  for (int nt = 0; nt < 16; ++nt) {
    const int n = nt * 16 + ncol;
    const float bias = out_b[n];
    v8f acc;
#pragma unroll
    for (int r = 0; r < 8; ++r) acc[r] = bias;

    const __bf16* wrow = out_wb + (size_t)n * DIM;
    if (nt < 15) __builtin_prefetch(out_wb + (size_t)(n + 16) * DIM, 0, 1);

#pragma unroll
    for (int ks = 0; ks < 8; ++ks) {
      const int k0 = ks * 32;
      union { v16bf v; v8bf h[2]; } A, B;
      const __bf16* ap = &s_xn[wave][m * DIM + k0 + khalf * 8];
      A.h[0] = *(const v8bf*)ap;
      A.h[1] = *(const v8bf*)(ap + 16);
      const __bf16* bp = wrow + k0 + khalf * 8;
      B.h[0] = *(const v8bf*)bp;
      B.h[1] = *(const v8bf*)(bp + 16);
      acc = __builtin_amdgcn_wmma_f32_16x16x32_bf16(
          false, A.v, false, B.v, (short)0, acc, false, false);
    }
#pragma unroll
    for (int r = 0; r < 8; ++r) {
      const int mr = (lane >> 4) * 8 + r;
      obuf[mr * DIM + n] = acc[r];            // ds_store_b32 (D layout)
    }
  }
  __syncthreads();

  // ---------- Phase 5: coalesced residual + store (float4) ----------
  {
    const float* ob   = obuf + m * DIM + khalf * 128;
    const float* xr   = x   + (size_t)(r0 + m) * DIM + khalf * 128;
    float*       outr = out + (size_t)(r0 + m) * DIM + khalf * 128;
    for (int j = 0; j < 128; j += 4) {
      float4 o = *(const float4*)(ob + j);    // ds_load_b128
      float4 v = *(const float4*)(xr + j);    // global_load_b128
      float4 rlt;
      rlt.x = o.x + v.x; rlt.y = o.y + v.y;
      rlt.z = o.z + v.z; rlt.w = o.w + v.w;
      *(float4*)(outr + j) = rlt;             // global_store_b128
    }
  }
}

extern "C" void kernel_launch(void* const* d_in, const int* in_sizes, int n_in,
                              void* d_out, int out_size, void* d_ws, size_t ws_size,
                              hipStream_t stream) {
  const float* x     = (const float*)d_in[0];
  const float* qkv_w = (const float*)d_in[1];
  const float* qkv_b = (const float*)d_in[2];
  const float* out_w = (const float*)d_in[3];
  const float* out_b = (const float*)d_in[4];
  const float* ln_g  = (const float*)d_in[5];
  const float* ln_b  = (const float*)d_in[6];
  float* out = (float*)d_out;

  __bf16* wb = (__bf16*)d_ws;                 // [768*256] qkv_w bf16, then [256*256] out_w bf16
  const int n_rows = in_sizes[0] / DIM;       // 131072

  const int cvt_total = DIM3 * DIM + DIM * DIM;
  hipLaunchKernelGGL(GA_wcvt_kernel, dim3((cvt_total + 255) / 256), dim3(256), 0, stream,
                     qkv_w, out_w, wb);
  hipLaunchKernelGGL(GA_attn_kernel, dim3(n_rows / RPB), dim3(WAVES * 32), 0, stream,
                     x, qkv_b, out_b, ln_g, ln_b,
                     wb, wb + (size_t)DIM3 * DIM, out);
}
